// FeedbackModelStudentFT_74311524155793
// MI455X (gfx1250) — matrix operand order repaired and checked by reading, not compiled, not run on hardware
//
#include <hip/hip_runtime.h>
#include <stdint.h>

// ---------------------------------------------------------------------------
// Types for CDNA5 WMMA (wave32): v_wmma_f32_16x16x32_bf16
// ---------------------------------------------------------------------------
typedef __attribute__((ext_vector_type(16))) __bf16 v16bf;
typedef __attribute__((ext_vector_type(8)))  float  v8f;

union AFrag {
    v16bf v;
    uint4 q[2];
};

static __device__ __forceinline__ unsigned short f2bf(float f) {
    uint32_t u = __float_as_uint(f);
    uint32_t r = u + 0x7FFFu + ((u >> 16) & 1u);   // round-to-nearest-even
    return (unsigned short)(r >> 16);
}
static __device__ __forceinline__ float bf2f(unsigned short h) {
    return __uint_as_float(((uint32_t)h) << 16);
}
static __device__ __forceinline__ float sigmoidf(float x) {
    return 1.0f / (1.0f + __expf(-x));
}

// ---------------------------------------------------------------------------
// Elementwise f32 -> bf16
// ---------------------------------------------------------------------------
__global__ __launch_bounds__(256) void fb_cvt_bf16(const float* __restrict__ in,
                                                   unsigned short* __restrict__ out,
                                                   size_t n) {
    size_t i = (size_t)blockIdx.x * 256 + threadIdx.x;
    if (i < n) out[i] = f2bf(in[i]);
}

// in[R][C] (f32) -> out[C][R] (bf16)
__global__ __launch_bounds__(256) void fb_transpose_bf16(const float* __restrict__ in,
                                                         unsigned short* __restrict__ out,
                                                         int R, int C) {
    size_t i = (size_t)blockIdx.x * 256 + threadIdx.x;
    if (i >= (size_t)R * C) return;
    int r = (int)(i / C);
    int c = (int)(i % C);
    out[(size_t)c * R + r] = f2bf(in[i]);
}

// ---------------------------------------------------------------------------
// Fragment loaders (CDNA5 16-bit A / B VGPR layouts)
// ---------------------------------------------------------------------------
static __device__ __forceinline__ void loadA(AFrag& a, const unsigned short* A,
                                             size_t rowbase, int k0, int half) {
    const unsigned short* ap = A + rowbase + k0 + half * 8;
    a.q[0] = *(const uint4*)(ap);
    a.q[1] = *(const uint4*)(ap + 16);
}
static __device__ __forceinline__ void loadB(AFrag& b, const unsigned short* Bt,
                                             int k0, int lane, int N, int n) {
    const int h = lane >> 4;
    const unsigned short* bp = Bt + (size_t)k0 * N + n + (lane & 15);
    unsigned short* e = (unsigned short*)&b;
#pragma unroll
    for (int s = 0; s < 16; ++s) {
        const int kk = (s < 8) ? (8 * h + s) : (16 + 8 * h + (s - 8));
        e[s] = bp[(size_t)kk * N];
    }
}

// ---------------------------------------------------------------------------
// Generic bf16 WMMA GEMM: C[m,n] = sum_k A[m,k]*Bt[k,n] (+bias_a[n])(+bias_b[n])
//                                  (+residual[m,n]);  f32 and/or bf16 output.
// A: [M][K] bf16 row-major;  Bt: [K][N] bf16 row-major (pre-transposed weight).
// M%16==0, N%64==0, K%64==0.  256 threads = 8 waves; each wave: 16x64 tile,
// ping-pong double-buffered fragments so loads overlap WMMA.
// ---------------------------------------------------------------------------
__global__ __launch_bounds__(256) void fb_wmma_gemm(
    const unsigned short* __restrict__ A,
    const unsigned short* __restrict__ Bt,
    const float* __restrict__ bias_a,
    const float* __restrict__ bias_b,
    const float* __restrict__ residual,
    float* __restrict__ Cf,
    unsigned short* __restrict__ Cbf,
    int M, int N, int K) {
    const int wave = threadIdx.x >> 5;
    const int lane = threadIdx.x & 31;
    const int tiles_n = N >> 6;
    const int total = (M >> 4) * tiles_n;
    const int tile = blockIdx.x * 8 + wave;
    if (tile >= total) return;       // wave-uniform branch: EXEC stays all-ones
    const int m0 = (tile / tiles_n) << 4;
    const int n0 = (tile % tiles_n) << 6;

    const int half = lane >> 4;
    const size_t arow = (size_t)(m0 + (lane & 15)) * K;

    v8f acc[4];
    acc[0] = {}; acc[1] = {}; acc[2] = {}; acc[3] = {};

    AFrag a0, a1;
    AFrag b0[4], b1[4];

    loadA(a0, A, arow, 0, half);
#pragma unroll
    for (int j = 0; j < 4; ++j) loadB(b0[j], Bt, 0, lane, N, n0 + 16 * j);

    for (int k0 = 0; k0 < K; k0 += 64) {
        // prefetch second half of this pair
        loadA(a1, A, arow, k0 + 32, half);
#pragma unroll
        for (int j = 0; j < 4; ++j) loadB(b1[j], Bt, k0 + 32, lane, N, n0 + 16 * j);

#pragma unroll
        for (int j = 0; j < 4; ++j)
            acc[j] = __builtin_amdgcn_wmma_f32_16x16x32_bf16(
                false, a0.v, false, b0[j].v, (short)0, acc[j], false, false);

        if (k0 + 64 < K) {
            loadA(a0, A, arow, k0 + 64, half);
#pragma unroll
            for (int j = 0; j < 4; ++j) loadB(b0[j], Bt, k0 + 64, lane, N, n0 + 16 * j);
        }

#pragma unroll
        for (int j = 0; j < 4; ++j)
            acc[j] = __builtin_amdgcn_wmma_f32_16x16x32_bf16(
                false, a1.v, false, b1[j].v, (short)0, acc[j], false, false);
    }

    const int nc = lane & 15;
#pragma unroll
    for (int j = 0; j < 4; ++j) {
        const int n = n0 + j * 16 + nc;
        float ba = bias_a ? bias_a[n] : 0.0f;
        float bb = bias_b ? bias_b[n] : 0.0f;
#pragma unroll
        for (int r = 0; r < 8; ++r) {
            const int m = m0 + r + half * 8;
            float vv = acc[j][r] + ba + bb;
            size_t idx = (size_t)m * N + n;
            if (residual) vv += residual[idx];
            if (Cf)  Cf[idx]  = vv;
            if (Cbf) Cbf[idx] = f2bf(vv);
        }
    }
}

// ---------------------------------------------------------------------------
// One LSTM time step:
//   gates[b, n] = xp[b, t, n] + sum_k h_prev[b,k] * w_hh[n,k]     (n in 0..4095)
// grid: 64 workgroups (16 hidden units each), 256 threads = 8 waves:
//   wave = (khalf<<2)|gate ; each wave does K=512 partial of one gate tile
//   (16 sequential WMMAs, ping-pong buffered), partials reduced in LDS.
// Activation fused; h written bf16 to hnext (rows 8..15 stay zero) + x_out.
// ---------------------------------------------------------------------------
__global__ __launch_bounds__(256) void fb_lstm_step(
    const unsigned short* __restrict__ hprev,   // [16][1024] bf16 (rows 8..15 = 0)
    const unsigned short* __restrict__ whh_t,   // [1024][4096] bf16
    const float* __restrict__ xp,               // [(B*S)][4096] f32, row = b*1024+t
    float* __restrict__ cstate,                 // [8][1024] f32
    unsigned short* __restrict__ hnext,         // [16][1024] bf16
    unsigned short* __restrict__ xout,          // [8][1024][1024] bf16
    int t) {
    __shared__ float gl[2][4][8][16];           // [khalf][gate][batch][unit]

    const int wave = threadIdx.x >> 5;
    const int lane = threadIdx.x & 31;
    const int gate = wave & 3;                  // 0=i 1=f 2=g 3=o
    const int kh   = wave >> 2;                 // K half
    const int u0 = blockIdx.x << 4;             // hidden-unit base
    const int n0 = gate * 1024 + u0;            // column in 4096-wide gate space

    const int half = lane >> 4;
    const size_t arow = (size_t)(lane & 15) * 1024;
    const int kbeg = kh * 512;

    v8f acc = {};
    AFrag a0, b0, a1, b1;
    loadA(a0, hprev, arow, kbeg, half);
    loadB(b0, whh_t, kbeg, lane, 4096, n0);

    for (int k0 = kbeg; k0 < kbeg + 512; k0 += 64) {
        loadA(a1, hprev, arow, k0 + 32, half);
        loadB(b1, whh_t, k0 + 32, lane, 4096, n0);
        acc = __builtin_amdgcn_wmma_f32_16x16x32_bf16(false, a0.v, false, b0.v,
                                                      (short)0, acc, false, false);
        if (k0 + 64 < kbeg + 512) {
            loadA(a0, hprev, arow, k0 + 64, half);
            loadB(b0, whh_t, k0 + 64, lane, 4096, n0);
        }
        acc = __builtin_amdgcn_wmma_f32_16x16x32_bf16(false, a1.v, false, b1.v,
                                                      (short)0, acc, false, false);
    }

    // rows 0..7 (= batch) live in half==0 lanes (VGPR r -> M=r)
    if (half == 0) {
        const int nc = lane & 15;
#pragma unroll
        for (int r = 0; r < 8; ++r) gl[kh][gate][r][nc] = acc[r];
    }
    __syncthreads();

    const int tid = threadIdx.x;
    if (tid < 128) {                            // (batch, unit) activation
        const int bb = tid >> 4;                // 0..7
        const int uu = tid & 15;                // 0..15
        const size_t xrow = ((size_t)bb * 1024 + t) * 4096;
        float gi = gl[0][0][bb][uu] + gl[1][0][bb][uu] + xp[xrow + 0 * 1024 + u0 + uu];
        float gf = gl[0][1][bb][uu] + gl[1][1][bb][uu] + xp[xrow + 1 * 1024 + u0 + uu];
        float gg = gl[0][2][bb][uu] + gl[1][2][bb][uu] + xp[xrow + 2 * 1024 + u0 + uu];
        float go = gl[0][3][bb][uu] + gl[1][3][bb][uu] + xp[xrow + 3 * 1024 + u0 + uu];
        const int cidx = bb * 1024 + u0 + uu;
        float cc = sigmoidf(gf) * cstate[cidx] + sigmoidf(gi) * tanhf(gg);
        cstate[cidx] = cc;
        float hh = sigmoidf(go) * tanhf(cc);
        unsigned short hb = f2bf(hh);
        hnext[cidx] = hb;
        xout[((size_t)bb * 1024 + t) * 1024 + u0 + uu] = hb;

        // prefetch next step's xp gate rows into cache (global_prefetch_b8)
        if (t + 1 < 1024) {
            const float* nx = xp + ((size_t)bb * 1024 + (t + 1)) * 4096 + u0 + uu;
            __builtin_prefetch(nx + 0 * 1024, 0, 0);
            __builtin_prefetch(nx + 1 * 1024, 0, 0);
            __builtin_prefetch(nx + 2 * 1024, 0, 0);
            __builtin_prefetch(nx + 3 * 1024, 0, 0);
        }
    }
}

// ---------------------------------------------------------------------------
// Ragged span-mean pooling: mean over s in (head, tail) exclusive.
// out f32 [8][32][1024];  x bf16 [8][1024][1024]
// ---------------------------------------------------------------------------
__global__ __launch_bounds__(256) void fb_pool(
    const unsigned short* __restrict__ x,
    const int* __restrict__ head, const int* __restrict__ tail,
    float* __restrict__ out) {
    size_t i = (size_t)blockIdx.x * 256 + threadIdx.x;   // 262144 total
    int h = (int)(i & 1023);
    int j = (int)((i >> 10) & 31);
    int b = (int)(i >> 15);
    int hd = head[b * 32 + j];
    int tl = tail[b * 32 + j];
    float s = 0.0f;
    for (int t = hd + 1; t < tl; ++t)
        s += bf2f(x[((size_t)b * 1024 + t) * 1024 + h]);
    out[i] = s / (float)(tl - hd - 1);
}

// ---------------------------------------------------------------------------
// LayerNorm over H=1024 (one row per 256-thread block); optional bf16 copy out
// ---------------------------------------------------------------------------
__global__ __launch_bounds__(256) void fb_layernorm(
    const float* __restrict__ in, const float* __restrict__ gamma,
    const float* __restrict__ beta, float* __restrict__ outf,
    unsigned short* __restrict__ outb) {
    __shared__ float red[256];
    const int row = blockIdx.x;
    const int tid = threadIdx.x;
    const float* x = in + (size_t)row * 1024;

    float s = 0.0f;
    for (int i = tid; i < 1024; i += 256) s += x[i];
    red[tid] = s; __syncthreads();
    for (int o = 128; o; o >>= 1) { if (tid < o) red[tid] += red[tid + o]; __syncthreads(); }
    float mu = red[0] / 1024.0f;
    __syncthreads();

    float v = 0.0f;
    for (int i = tid; i < 1024; i += 256) { float d = x[i] - mu; v += d * d; }
    red[tid] = v; __syncthreads();
    for (int o = 128; o; o >>= 1) { if (tid < o) red[tid] += red[tid + o]; __syncthreads(); }
    float inv = rsqrtf(red[0] / 1024.0f + 1e-7f);

    for (int i = tid; i < 1024; i += 256) {
        float y = (x[i] - mu) * inv * gamma[i] + beta[i];
        outf[(size_t)row * 1024 + i] = y;
        if (outb) outb[(size_t)row * 1024 + i] = f2bf(y);
    }
}

// ---------------------------------------------------------------------------
// MHA over J=32 spans; one wave32 block per (batch, head); d=64, 16 heads.
// ---------------------------------------------------------------------------
__global__ __launch_bounds__(32) void fb_attn(
    const float* __restrict__ q, const float* __restrict__ k,
    const float* __restrict__ v, const int* __restrict__ mask,
    float* __restrict__ ctx) {
    const int b = blockIdx.x >> 4;
    const int h = blockIdx.x & 15;
    const int j = threadIdx.x;
    const float NEGMAX = -3.402823466e38f;

    const float* qr = q + ((size_t)(b * 32 + j)) * 1024 + h * 64;
    const int mj = mask[b * 32 + j];

    float sc[32];
    float mx = NEGMAX;
#pragma unroll
    for (int kk = 0; kk < 32; ++kk) {
        const float* kr = k + ((size_t)(b * 32 + kk)) * 1024 + h * 64;
        float s = 0.0f;
        for (int d = 0; d < 64; ++d) s += qr[d] * kr[d];
        s *= 0.125f;                                   // 1/sqrt(64)
        bool valid = (mj != 0) && (mask[b * 32 + kk] != 0);
        sc[kk] = valid ? s : NEGMAX;
        mx = fmaxf(mx, sc[kk]);
    }
    float ssum = 0.0f;
#pragma unroll
    for (int kk = 0; kk < 32; ++kk) { float e = __expf(sc[kk] - mx); sc[kk] = e; ssum += e; }
    float inv = 1.0f / ssum;
#pragma unroll
    for (int kk = 0; kk < 32; ++kk) {
        bool valid = (mj != 0) && (mask[b * 32 + kk] != 0);
        sc[kk] = valid ? sc[kk] * inv : 0.0f;          // XSoftmax semantics
    }
    float* cr = ctx + ((size_t)(b * 32 + j)) * 1024 + h * 64;
    for (int d = 0; d < 64; ++d) {
        float s = 0.0f;
#pragma unroll
        for (int kk = 0; kk < 32; ++kk)
            s += sc[kk] * v[((size_t)(b * 32 + kk)) * 1024 + h * 64 + d];
        cr[d] = s;
    }
}

// ---------------------------------------------------------------------------
// Classifier: logits[row, c] = fv2[row,:] . cls_w[c,:] + cls_b[c]   (768 thr)
// ---------------------------------------------------------------------------
__global__ __launch_bounds__(256) void fb_classifier(
    const float* __restrict__ fv2, const float* __restrict__ w,
    const float* __restrict__ bias, float* __restrict__ out) {
    int gid = blockIdx.x * 256 + threadIdx.x;
    if (gid >= 768) return;
    int row = gid / 3, c = gid % 3;
    float s = bias[c];
    const float* xr = fv2 + (size_t)row * 1024;
    const float* wr = w + (size_t)c * 1024;
    for (int kk = 0; kk < 1024; ++kk) s += xr[kk] * wr[kk];
    out[gid] = s;
}

// ---------------------------------------------------------------------------
// CE (ignore_index=-1) + focal (gamma=3); 256 rows, 3 classes; one block.
// ---------------------------------------------------------------------------
__global__ __launch_bounds__(256) void fb_loss(
    const float* __restrict__ logits, const int* __restrict__ labels,
    float* __restrict__ out_loss) {
    __shared__ float sn[256], sf[256], sv[256];
    const int t = threadIdx.x;
    float l0 = logits[t * 3 + 0], l1 = logits[t * 3 + 1], l2 = logits[t * 3 + 2];
    int lab = labels[t];
    float vf = (lab >= 0) ? 1.0f : 0.0f;
    int lc = lab < 0 ? 0 : (lab > 2 ? 2 : lab);
    float m = fmaxf(l0, fmaxf(l1, l2));
    float e0 = __expf(l0 - m), e1 = __expf(l1 - m), e2 = __expf(l2 - m);
    float lse = m + __logf(e0 + e1 + e2);
    float lt = (lc == 0) ? l0 : (lc == 1) ? l1 : l2;
    float logp = lt - lse;
    float pt = __expf(logp);
    float om = 1.0f - pt;
    sn[t] = (-logp) * vf;
    sf[t] = (-(om * om * om * logp)) * vf;
    sv[t] = vf;
    __syncthreads();
    for (int o = 128; o; o >>= 1) {
        if (t < o) { sn[t] += sn[t + o]; sf[t] += sf[t + o]; sv[t] += sv[t + o]; }
        __syncthreads();
    }
    if (t == 0) out_loss[0] = sn[0] / sv[0] + sf[0] / sv[0];
}

// ---------------------------------------------------------------------------
// Host-side orchestration
// ---------------------------------------------------------------------------
extern "C" void kernel_launch(void* const* d_in, const int* in_sizes, int n_in,
                              void* d_out, int out_size, void* d_ws, size_t ws_size,
                              hipStream_t stream) {
    (void)in_sizes; (void)n_in; (void)out_size; (void)ws_size;

    const float* hidden    = (const float*)d_in[0];
    const int*   span_head = (const int*)  d_in[1];
    const int*   span_tail = (const int*)  d_in[2];
    const int*   span_mask = (const int*)  d_in[3];
    const int*   labels    = (const int*)  d_in[4];
    const float* w_ih      = (const float*)d_in[5];
    const float* w_hh      = (const float*)d_in[6];
    const float* b_ih      = (const float*)d_in[7];
    const float* b_hh      = (const float*)d_in[8];
    const float* ln_g      = (const float*)d_in[9];
    const float* ln_b      = (const float*)d_in[10];
    const float* wq = (const float*)d_in[11]; const float* bq = (const float*)d_in[12];
    const float* wk = (const float*)d_in[13]; const float* bk = (const float*)d_in[14];
    const float* wv = (const float*)d_in[15]; const float* bv = (const float*)d_in[16];
    const float* wo = (const float*)d_in[17]; const float* bo = (const float*)d_in[18];
    const float* ln2_g = (const float*)d_in[19];
    const float* ln2_b = (const float*)d_in[20];
    const float* cls_w = (const float*)d_in[21];
    const float* cls_b = (const float*)d_in[22];

    const int B = 8, S = 1024, H = 1024, J = 32;
    const size_t BS = (size_t)B * S;          // 8192 rows

    // workspace carve-out
    char* wsc = (char*)d_ws;
    size_t off = 0;
    auto alloc = [&](size_t bytes) -> void* {
        off = (off + 255) & ~(size_t)255;
        void* p = wsc + off;
        off += bytes;
        return p;
    };
    unsigned short* hidden_bf = (unsigned short*)alloc(BS * H * 2);           // 16.8 MB
    unsigned short* w_ih_t    = (unsigned short*)alloc((size_t)H * 4 * H * 2);// 8.4 MB
    unsigned short* w_hh_t    = (unsigned short*)alloc((size_t)H * 4 * H * 2);// 8.4 MB
    unsigned short* wq_t      = (unsigned short*)alloc((size_t)H * H * 2);
    unsigned short* wk_t      = (unsigned short*)alloc((size_t)H * H * 2);
    unsigned short* wv_t      = (unsigned short*)alloc((size_t)H * H * 2);
    unsigned short* wo_t      = (unsigned short*)alloc((size_t)H * H * 2);
    float*          xp        = (float*)alloc(BS * 4 * H * 4);                // 134 MB
    unsigned short* x_out     = (unsigned short*)alloc(BS * H * 2);           // 16.8 MB
    unsigned short* h_buf     = (unsigned short*)alloc(2 * 16 * (size_t)H * 2);
    float*          c_state   = (float*)alloc((size_t)B * H * 4);
    float*          span_vec  = (float*)alloc((size_t)B * J * H * 4);         // 1 MB
    float*          fv        = (float*)alloc((size_t)B * J * H * 4);
    unsigned short* fv_bf     = (unsigned short*)alloc((size_t)B * J * H * 2);
    float*          qbuf      = (float*)alloc((size_t)B * J * H * 4);
    float*          kbuf      = (float*)alloc((size_t)B * J * H * 4);
    float*          vbuf      = (float*)alloc((size_t)B * J * H * 4);
    float*          ctx       = (float*)alloc((size_t)B * J * H * 4);
    unsigned short* ctx_bf    = (unsigned short*)alloc((size_t)B * J * H * 2);
    float*          attn_out  = (float*)alloc((size_t)B * J * H * 4);
    float*          fv2       = (float*)alloc((size_t)B * J * H * 4);

    float* logits = (float*)d_out;            // [256][3]
    float* loss   = (float*)d_out + 768;

    // --- prep: conversions / transposes / state init ---
    {
        size_t n = BS * H;
        fb_cvt_bf16<<<(int)((n + 255) / 256), 256, 0, stream>>>(hidden, hidden_bf, n);
    }
    fb_transpose_bf16<<<(4096 * 1024 + 255) / 256, 256, 0, stream>>>(w_ih, w_ih_t, 4096, 1024);
    fb_transpose_bf16<<<(4096 * 1024 + 255) / 256, 256, 0, stream>>>(w_hh, w_hh_t, 4096, 1024);
    fb_transpose_bf16<<<(1024 * 1024 + 255) / 256, 256, 0, stream>>>(wq, wq_t, 1024, 1024);
    fb_transpose_bf16<<<(1024 * 1024 + 255) / 256, 256, 0, stream>>>(wk, wk_t, 1024, 1024);
    fb_transpose_bf16<<<(1024 * 1024 + 255) / 256, 256, 0, stream>>>(wv, wv_t, 1024, 1024);
    fb_transpose_bf16<<<(1024 * 1024 + 255) / 256, 256, 0, stream>>>(wo, wo_t, 1024, 1024);
    hipMemsetAsync(h_buf, 0, 2 * 16 * (size_t)H * 2, stream);
    hipMemsetAsync(c_state, 0, (size_t)B * H * 4, stream);

    // --- LSTM input projection: xp = hidden @ w_ih^T + b_ih + b_hh ---
    {
        int tiles = (8192 / 16) * (4096 / 64);     // 32768 wave-tiles
        fb_wmma_gemm<<<tiles / 8, 256, 0, stream>>>(
            hidden_bf, w_ih_t, b_ih, b_hh, nullptr, xp, nullptr, 8192, 4096, 1024);
    }

    // --- LSTM recurrence (sequential scan, double-buffered h) ---
    for (int t = 0; t < S; ++t) {
        const unsigned short* hp = h_buf + (size_t)(t & 1) * 16 * H;
        unsigned short*       hn = h_buf + (size_t)((t + 1) & 1) * 16 * H;
        fb_lstm_step<<<64, 256, 0, stream>>>(hp, w_hh_t, xp, c_state, hn, x_out, t);
    }

    // --- span-mean pooling + LN ---
    fb_pool<<<1024, 256, 0, stream>>>(x_out, span_head, span_tail, span_vec);
    fb_layernorm<<<256, 256, 0, stream>>>(span_vec, ln_g, ln_b, fv, fv_bf);

    // --- q/k/v projections (M=256, N=1024, K=1024) ---
    {
        int tiles = (256 / 16) * (1024 / 64);      // 256 wave-tiles -> 32 blocks
        fb_wmma_gemm<<<tiles / 8, 256, 0, stream>>>(fv_bf, wq_t, bq, nullptr, nullptr, qbuf, nullptr, 256, 1024, 1024);
        fb_wmma_gemm<<<tiles / 8, 256, 0, stream>>>(fv_bf, wk_t, bk, nullptr, nullptr, kbuf, nullptr, 256, 1024, 1024);
        fb_wmma_gemm<<<tiles / 8, 256, 0, stream>>>(fv_bf, wv_t, bv, nullptr, nullptr, vbuf, nullptr, 256, 1024, 1024);
    }

    // --- attention over 32 spans ---
    fb_attn<<<B * 16, 32, 0, stream>>>(qbuf, kbuf, vbuf, span_mask, ctx);

    // --- output projection + residual, then LN2 ---
    {
        size_t n = (size_t)B * J * H;
        fb_cvt_bf16<<<(int)((n + 255) / 256), 256, 0, stream>>>(ctx, ctx_bf, n);
        int tiles = (256 / 16) * (1024 / 64);
        fb_wmma_gemm<<<tiles / 8, 256, 0, stream>>>(ctx_bf, wo_t, bo, nullptr, fv, attn_out, nullptr, 256, 1024, 1024);
    }
    fb_layernorm<<<256, 256, 0, stream>>>(attn_out, ln2_g, ln2_b, fv2, nullptr);

    // --- classifier + loss ---
    fb_classifier<<<3, 256, 0, stream>>>(fv2, cls_w, cls_b, logits);
    fb_loss<<<1, 256, 0, stream>>>(logits, labels, loss);
}
